// SimTransform_10471130268241
// MI455X (gfx1250) — matrix operand order repaired and checked by reading, not compiled, not run on hardware
//
#include <hip/hip_runtime.h>

// ---------------------------------------------------------------------------
// SimTransform for MI455X (gfx1250, wave32):
//   conv3x3 s2 (512->128) + BN + ReLU  -> implicit GEMM, v_wmma_f32_16x16x32_bf16
//     fast path: channels-last padded bf16 input + async global->LDS staging
//   L2-normalize over channels         -> VALU, emitted on a zero-padded halo grid
//   windowed correlation (5x9x9)       -> per-tile GEMM, v_wmma_f32_16x16x32_bf16
// Identity used: post[b,t,l] = y[b,t+l-2]  =>  f2[b,t,l] == f1[b,t+l-2]
// ---------------------------------------------------------------------------

typedef __bf16 bf16_t;
typedef __attribute__((ext_vector_type(8)))  bf16_t bf16x8;
typedef __attribute__((ext_vector_type(16))) bf16_t bf16x16;
typedef __attribute__((ext_vector_type(8)))  float  f32x8;

// async global->LDS b64 payload type per the builtin's signature:
//   (int __vector(2) addrspace(1)*, int __vector(2) addrspace(3)*, Ii, Ii)
typedef int v2i_vs __attribute__((vector_size(8)));
typedef __attribute__((address_space(1))) v2i_vs* gptr_v2i;
typedef __attribute__((address_space(3))) v2i_vs* lptr_v2i;

#define NIMG 32            // B*T
#define CIN  512
#define HIN  56
#define HINP 58            // padded input extent (56 + 1 halo each side)
#define COUT 128
#define HOUT 28
#define NPIX (HOUT*HOUT)   // 784
#define KRED (CIN*9)       // 4608 reduction length for the conv GEMM
#define NT   8             // temporal segments
#define NL   5             // temporal window
#define HP   36            // padded spatial extent for f1 (28 + 2*4 halo)
#define NPP  (HP*HP)       // 1296 padded pixels

#if __has_builtin(__builtin_amdgcn_global_load_async_to_lds_b64) && \
    __has_builtin(__builtin_amdgcn_s_wait_asynccnt)
#define USE_ASYNC_LDS 1
#else
#define USE_ASYNC_LDS 0
#endif

__device__ __forceinline__ bf16_t f2bf(float f) { return (bf16_t)f; }

// ---------------------------------------------------------------------------
// Kernel 1: weights -> bf16 GEMM layout  wA[oc][(kh*3+kw)*512 + c],
//           fold BN into scale/bias.
// ---------------------------------------------------------------------------
__global__ void prep_kernel(const float* __restrict__ w,
                            const float* __restrict__ gamma,
                            const float* __restrict__ beta,
                            const float* __restrict__ mean,
                            const float* __restrict__ var,
                            bf16_t* __restrict__ wA,
                            float* __restrict__ scale,
                            float* __restrict__ bias) {
  int idx = blockIdx.x * blockDim.x + threadIdx.x;
  if (idx < COUT) {
    float s = gamma[idx] * rsqrtf(var[idx] + 1e-5f);
    scale[idx] = s;
    bias[idx]  = beta[idx] - mean[idx] * s;
  }
  if (idx < COUT * KRED) {
    int oc = idx / KRED, r = idx % KRED;
    int kpos = r / CIN, c = r % CIN;            // r = kpos*512 + c
    wA[idx] = f2bf(w[oc * KRED + c * 9 + kpos]);
  }
}

// ---------------------------------------------------------------------------
// Kernel 1b (fast path): transpose + pad x into channels-last bf16:
//   xt[img][ihp 0..57][iwp 0..57][c 0..511], zero in the 1-wide border.
//   LDS-tiled so both global sides are coalesced.
//   grid: img * 58 * 16 c-blocks of 32 channels.
// ---------------------------------------------------------------------------
__global__ void __launch_bounds__(256)
xpad_kernel(const float* __restrict__ x, bf16_t* __restrict__ xt) {
  __shared__ float sT[32][65];                 // [c][iw], padded vs bank conflicts
  const int bid  = blockIdx.x;
  const int cblk = bid & 15;
  const int q    = bid >> 4;
  const int ihp  = q % HINP;
  const int img  = q / HINP;
  const int tid  = threadIdx.x;
  const bool interior = (ihp >= 1) && (ihp <= HIN);

  if (interior) {
    const int ih = ihp - 1;
    const float* xrow = x + (size_t)(img * CIN + cblk * 32) * (HIN * HIN) + ih * HIN;
#pragma unroll
    for (int r = 0; r < 8; ++r) {              // 2048 = 32c x 64iw slots
      const int idx = r * 256 + tid;
      const int c = idx >> 6, iw = idx & 63;
      sT[c][iw] = (iw < HIN) ? xrow[(size_t)c * (HIN * HIN) + iw] : 0.f;
    }
  }
  __syncthreads();

  bf16_t* orow = xt + ((size_t)(img * HINP + ihp) * HINP) * CIN + cblk * 32;
#pragma unroll
  for (int r = 0; r < 8; ++r) {                // 1856 = 58iwp x 32c
    const int idx = r * 256 + tid;
    if (idx < HINP * 32) {
      const int iwp = idx >> 5, c = idx & 31;
      float v = 0.f;
      if (interior && iwp >= 1 && iwp <= HIN) v = sT[c][iwp - 1];
      orow[(size_t)iwp * CIN + c] = f2bf(v);
    }
  }
}

// ---------------------------------------------------------------------------
// Kernel 2 (fast path): conv + BN + ReLU as implicit GEMM from padded
// channels-last bf16 input.
//   grid.x = img*28 (one output row per workgroup), 8 waves.
//   Wave w: oc tile [16w,16w+16), N = 32 pixels (28 real), K = 4608 in 144
//   chunks of 32; B chunk staged in LDS with one async global->LDS B64 copy
//   per thread (no conversion, no bounds checks: padding is in xt).
// ---------------------------------------------------------------------------
__global__ void __launch_bounds__(256)
conv_fast_kernel(const bf16_t* __restrict__ xt,
                 const bf16_t* __restrict__ wA,
                 const float* __restrict__ scale,
                 const float* __restrict__ bias,
                 float* __restrict__ y) {
  __shared__ alignas(64) bf16_t sBt[2][32][32];   // [buf][ow][c_local]

  const int img  = blockIdx.x / HOUT;
  const int oh   = blockIdx.x % HOUT;
  const int tid  = threadIdx.x;
  const int wv   = tid >> 5;
  const int lane = tid & 31;
  const int half = lane >> 4;
  const int l16  = lane & 15;

  // staging: thread -> (ow, 8B segment of the 64B channel run)
  const int ow_s = tid >> 3;            // 0..31
  const int seg  = tid & 7;             // 0..7 -> 4 channels each

  const bf16_t* xti = xt + (size_t)img * (HINP * HINP) * CIN;

  auto stage = [&](int kc, int buf) {
    const int kpos = kc >> 4;           // 0..8  (kh*3+kw)
    const int c0   = (kc & 15) << 5;    // 0..480
    const int kh = kpos / 3, kw = kpos % 3;
    const int ih1 = 2 * oh + kh;        // 0..56, always in range (pad in xt)
    int iw1 = 2 * ow_s + kw;            // pad pixels ow>=28 clamp onto zero col
    iw1 = iw1 > HINP - 1 ? HINP - 1 : iw1;
    const bf16_t* g = xti + (size_t)(ih1 * HINP + iw1) * CIN + c0 + seg * 4;
    bf16_t* l = &sBt[buf][ow_s][seg * 4];
#if USE_ASYNC_LDS
    __builtin_amdgcn_global_load_async_to_lds_b64(
        (gptr_v2i)(const void*)g, (lptr_v2i)(void*)l, 0, 0);
#else
    *(uint2*)l = *(const uint2*)g;
#endif
  };

  f32x8 acc0 = {};   // pixels 0..15
  f32x8 acc1 = {};   // pixels 16..31 (28..31 are pad, never stored)

  stage(0, 0);
#if USE_ASYNC_LDS
  __builtin_amdgcn_s_wait_asynccnt(0);
#endif
  __syncthreads();

  const int arow = (16 * wv + l16) * KRED;
  const int kb8  = half * 16;

#pragma unroll 2
  for (int kc = 0; kc < 144; ++kc) {
    const int buf = kc & 1;
    if (kc + 1 < 144) stage(kc + 1, buf ^ 1);
    if (kc + 2 < 144)                       // gfx1250 global_prefetch_b8
      __builtin_prefetch(wA + arow + (kc + 2) * 32 + kb8, 0, 3);

    const bf16_t* ap = wA + arow + kc * 32 + kb8;
    bf16x8 alo = *(const bf16x8*)ap;
    bf16x8 ahi = *(const bf16x8*)(ap + 8);
    bf16x16 a = __builtin_shufflevector(alo, ahi,
        0,1,2,3,4,5,6,7,8,9,10,11,12,13,14,15);

    bf16x16 b0 = *(const bf16x16*)&sBt[buf][l16][half * 16];
    bf16x16 b1 = *(const bf16x16*)&sBt[buf][16 + l16][half * 16];

    acc0 = __builtin_amdgcn_wmma_f32_16x16x32_bf16(false, a, false, b0,
                                                   (short)0, acc0, false, false);
    acc1 = __builtin_amdgcn_wmma_f32_16x16x32_bf16(false, a, false, b1,
                                                   (short)0, acc1, false, false);
#if USE_ASYNC_LDS
    __builtin_amdgcn_s_wait_asynccnt(0);
#endif
    __syncthreads();
  }

  float* yrow = y + (size_t)img * COUT * NPIX + oh * HOUT;
#pragma unroll
  for (int r = 0; r < 8; ++r) {
    const int oc = 16 * wv + 8 * half + r;
    const float s = scale[oc], bb = bias[oc];
    yrow[(size_t)oc * NPIX + l16] = fmaxf(acc0[r] * s + bb, 0.f);
    const int ow1 = 16 + l16;
    if (ow1 < HOUT)
      yrow[(size_t)oc * NPIX + ow1] = fmaxf(acc1[r] * s + bb, 0.f);
  }
}

// ---------------------------------------------------------------------------
// Kernel 2 (fallback path, small workspace): conv + BN + ReLU straight from
// f32 NCHW input with predicated im2col staging. Same math as fast path.
// ---------------------------------------------------------------------------
__global__ void __launch_bounds__(256)
conv_bn_relu_kernel(const float* __restrict__ x,
                    const bf16_t* __restrict__ wA,
                    const float* __restrict__ scale,
                    const float* __restrict__ bias,
                    float* __restrict__ y) {
  __shared__ alignas(64) bf16_t sBt[2][32][32];

  const int img  = blockIdx.x / HOUT;
  const int oh   = blockIdx.x % HOUT;
  const int tid  = threadIdx.x;
  const int wv   = tid >> 5;
  const int lane = tid & 31;
  const int half = lane >> 4;
  const int l16  = lane & 15;

  const float* ximg = x + (size_t)img * CIN * HIN * HIN;
  const int lc   = tid >> 3;
  const int low4 = (tid & 7) * 4;

  auto load_chunk = [&](int kc, int buf) {
    const int kpos = kc >> 4;
    const int c0   = (kc & 15) << 5;
    const int kh = kpos / 3, kw = kpos % 3;
    const int ih = 2 * oh + kh - 1;
    if (ih >= 0 && ih < HIN) {
      const float* row = ximg + (size_t)(c0 + lc) * (HIN * HIN) + ih * HIN;
#pragma unroll
      for (int j = 0; j < 4; ++j) {
        const int ow = low4 + j;
        const int iw = 2 * ow + kw - 1;
        int iwc = iw < 0 ? 0 : (iw > HIN - 1 ? HIN - 1 : iw);
        float v = row[iwc];
        const bool ok = (iw >= 0) & (iw < HIN) & (ow < HOUT);
        sBt[buf][ow][lc] = f2bf(ok ? v : 0.f);
      }
    } else {
#pragma unroll
      for (int j = 0; j < 4; ++j) sBt[buf][low4 + j][lc] = (bf16_t)0.f;
    }
  };

  f32x8 acc0 = {};
  f32x8 acc1 = {};

  load_chunk(0, 0);
  __syncthreads();

  const int arow = (16 * wv + l16) * KRED;
  const int kb8  = half * 16;

#pragma unroll 2
  for (int kc = 0; kc < 144; ++kc) {
    const int buf = kc & 1;
    if (kc + 1 < 144) load_chunk(kc + 1, buf ^ 1);

    const bf16_t* ap = wA + arow + kc * 32 + kb8;
    bf16x8 alo = *(const bf16x8*)ap;
    bf16x8 ahi = *(const bf16x8*)(ap + 8);
    bf16x16 a = __builtin_shufflevector(alo, ahi,
        0,1,2,3,4,5,6,7,8,9,10,11,12,13,14,15);

    bf16x16 b0 = *(const bf16x16*)&sBt[buf][l16][half * 16];
    bf16x16 b1 = *(const bf16x16*)&sBt[buf][16 + l16][half * 16];

    acc0 = __builtin_amdgcn_wmma_f32_16x16x32_bf16(false, a, false, b0,
                                                   (short)0, acc0, false, false);
    acc1 = __builtin_amdgcn_wmma_f32_16x16x32_bf16(false, a, false, b1,
                                                   (short)0, acc1, false, false);
    __syncthreads();
  }

  float* yrow = y + (size_t)img * COUT * NPIX + oh * HOUT;
#pragma unroll
  for (int r = 0; r < 8; ++r) {
    const int oc = 16 * wv + 8 * half + r;
    const float s = scale[oc], bb = bias[oc];
    yrow[(size_t)oc * NPIX + l16] = fmaxf(acc0[r] * s + bb, 0.f);
    const int ow1 = 16 + l16;
    if (ow1 < HOUT)
      yrow[(size_t)oc * NPIX + ow1] = fmaxf(acc1[r] * s + bb, 0.f);
  }
}

// ---------------------------------------------------------------------------
// Kernel 3: L2 normalize over channels; emit bf16 in [img][padded pixel][ch]
// layout on a 36x36 halo grid (zeros in the 4-wide border) so the correlation
// kernel needs no bounds checks at all.
// ---------------------------------------------------------------------------
__global__ void norm_kernel(const float* __restrict__ y,
                            bf16_t* __restrict__ f1) {
  const int p = blockIdx.x * blockDim.x + threadIdx.x;
  if (p >= NIMG * NPP) return;
  const int img = p / NPP, pp = p % NPP;
  const int pi = pp / HP - 4, pj = pp % HP - 4;
  bf16_t* fp = f1 + (size_t)p * COUT;
  if (pi < 0 || pi >= HOUT || pj < 0 || pj >= HOUT) {
    uint4* q = (uint4*)fp;
#pragma unroll
    for (int i = 0; i < 16; ++i) q[i] = make_uint4(0u, 0u, 0u, 0u);
    return;
  }
  const float* yp = y + (size_t)img * COUT * NPIX + pi * HOUT + pj;
  float ss = 0.f;
#pragma unroll 4
  for (int c = 0; c < COUT; ++c) { float v = yp[(size_t)c * NPIX]; ss += v * v; }
  const float inv = rsqrtf(ss + 1e-6f);
#pragma unroll 4
  for (int c = 0; c < COUT; ++c) fp[c] = f2bf(yp[(size_t)c * NPIX] * inv);
}

// ---------------------------------------------------------------------------
// Kernel 4: windowed correlation.
//   One wave per job (b,t,l, 4x4-pixel tile): A = 16 pixels x 128ch (frame t),
//   B = 128ch x 144 halo pixels (frame t+l-2, 12x12 window on the padded
//   grid, so no bounds checks), C = 16x144 via 9 N-tiles x 4 K-chunks =
//   36 WMMAs; scatter the 81 valid (u,v) per pixel.
// ---------------------------------------------------------------------------
__global__ void __launch_bounds__(256)
corr_kernel(const bf16_t* __restrict__ f1, float* __restrict__ out) {
  const int wv   = threadIdx.x >> 5;
  const int lane = threadIdx.x & 31;
  const int job  = blockIdx.x * 8 + wv;          // 980*8 == 7840 jobs exactly

  const int tile = job % 49; int rest = job / 49;
  const int l = rest % NL;   rest /= NL;
  const int t = rest % NT;
  const int b = rest / NT;
  const int pi0 = (tile / 7) * 4, pj0 = (tile % 7) * 4;
  const int half = lane >> 4, l16 = lane & 15;
  const int tp = t + l - 2;

  if (tp < 0 || tp >= NT) {                      // wave-uniform zero path
    for (int m = 0; m < 16; ++m) {
      const int pi = pi0 + (m >> 2), pj = pj0 + (m & 3);
      float* o = out + ((size_t)((b * NT + t) * HOUT + pi) * HOUT + pj) * 405
                     + l * 81;
      for (int e = lane; e < 81; e += 32) o[e] = 0.f;
    }
    return;
  }

  const bf16_t* fA = f1 + (size_t)(b * NT + t)  * NPP * COUT;
  const bf16_t* fB = f1 + (size_t)(b * NT + tp) * NPP * COUT;

  const int am   = l16;
  const int apix = (pi0 + (am >> 2) + 4) * HP + (pj0 + (am & 3) + 4);
  const bf16_t* aprow = fA + (size_t)apix * COUT + half * 16;

  const bf16_t* brow[9];
#pragma unroll
  for (int nt = 0; nt < 9; ++nt) {
    const int q  = nt * 16 + l16;
    const int qi = q / 12, qj = q % 12;
    brow[nt] = fB + (size_t)((pi0 + qi) * HP + (pj0 + qj)) * COUT + half * 16;
  }

  f32x8 acc[9] = {};
  for (int kc = 0; kc < 4; ++kc) {
    const bf16_t* ap = aprow + kc * 32;
    bf16x8 alo = *(const bf16x8*)ap;
    bf16x8 ahi = *(const bf16x8*)(ap + 8);
    bf16x16 a = __builtin_shufflevector(alo, ahi,
        0,1,2,3,4,5,6,7,8,9,10,11,12,13,14,15);
#pragma unroll
    for (int nt = 0; nt < 9; ++nt) {
      bf16x16 bv = *(const bf16x16*)(brow[nt] + kc * 32);
      acc[nt] = __builtin_amdgcn_wmma_f32_16x16x32_bf16(false, a, false, bv,
                                                        (short)0, acc[nt],
                                                        false, false);
    }
  }

#pragma unroll
  for (int nt = 0; nt < 9; ++nt) {
    const int q  = nt * 16 + l16;
    const int qi = q / 12, qj = q % 12;
#pragma unroll
    for (int r = 0; r < 8; ++r) {
      const int m  = 8 * half + r;
      const int mi = m >> 2, mj = m & 3;
      const int u = qi - mi, v = qj - mj;
      if (u >= 0 && u < 9 && v >= 0 && v < 9) {
        const int pi = pi0 + mi, pj = pj0 + mj;
        out[((size_t)((b * NT + t) * HOUT + pi) * HOUT + pj) * 405
            + l * 81 + u * 9 + v] = acc[nt][r];
      }
    }
  }
}

// ---------------------------------------------------------------------------
extern "C" void kernel_launch(void* const* d_in, const int* in_sizes, int n_in,
                              void* d_out, int out_size, void* d_ws,
                              size_t ws_size, hipStream_t stream) {
  (void)in_sizes; (void)n_in; (void)out_size;
  const float* x     = (const float*)d_in[0];
  const float* w     = (const float*)d_in[1];
  const float* gamma = (const float*)d_in[2];
  const float* beta  = (const float*)d_in[3];
  const float* mean  = (const float*)d_in[4];
  const float* var   = (const float*)d_in[5];
  float* out = (float*)d_out;
  char*  ws  = (char*)d_ws;

  // workspace layout (all 512B aligned)
  bf16_t* wA    = (bf16_t*)(ws);                 // 128*4608 bf16 = 1,179,648 B
  float*  scale = (float*)(ws + 1179648);        // 512 B
  float*  bias  = (float*)(ws + 1180160);        // 512 B
  float*  y     = (float*)(ws + 1180672);        // 32*128*784 f32 = 12,845,056 B
  bf16_t* f1    = (bf16_t*)(ws + 14025728);      // 32*1296*128 bf16 = 10,616,832 B
  bf16_t* xt    = (bf16_t*)(ws + 24642560);      // 32*58*58*512 bf16 = 110,231,552 B
  const size_t NEED_FAST = 24642560ull + 110231552ull;   // 134,874,112 B

  prep_kernel<<<(COUT * KRED + 255) / 256, 256, 0, stream>>>(
      w, gamma, beta, mean, var, wA, scale, bias);

  if (ws_size >= NEED_FAST) {
    xpad_kernel<<<NIMG * HINP * 16, 256, 0, stream>>>(x, xt);
    conv_fast_kernel<<<NIMG * HOUT, 256, 0, stream>>>(xt, wA, scale, bias, y);
  } else {
    conv_bn_relu_kernel<<<NIMG * HOUT, 256, 0, stream>>>(x, wA, scale, bias, y);
  }

  norm_kernel<<<(NIMG * NPP + 255) / 256, 256, 0, stream>>>(y, f1);
  corr_kernel<<<980, 256, 0, stream>>>(f1, out);
}
